// GrainGAT_20375324852412
// MI455X (gfx1250) — matrix operand order repaired and checked by reading, not compiled, not run on hardware
//
#include <hip/hip_runtime.h>
#include <hip/hip_bf16.h>

// ---------------------------------------------------------------------------
// GrainGAT on gfx1250: 2-layer GAT.
//   Layer1: h1 = x(100000x128) @ W1(128x64)  -> bf16 WMMA 16x16x32, f32 acc.
//           B fragments register-resident; A staged in LDS pre-swizzled into
//           per-lane fragment order with vector (b128) staging both ways.
//   Edge softmax + aggregation via L2-resident atomics (arrays ~25MB << 192MB L2)
//   Layer2: tiny 64->2 projection, scalar edge passes.
// ---------------------------------------------------------------------------

typedef __attribute__((ext_vector_type(16))) __bf16          v16bf;
typedef __attribute__((ext_vector_type(16))) unsigned short  v16us;
typedef __attribute__((ext_vector_type(8)))  float           v8f;

#define DIN  128
#define DOUT 64
#define NEG_SLOPE 0.2f

static __device__ __forceinline__ unsigned short f2bf(float f) {
    unsigned int u = __float_as_uint(f);
    u += 0x7FFFu + ((u >> 16) & 1u);     // round-to-nearest-even
    return (unsigned short)(u >> 16);
}
static __device__ __forceinline__ unsigned int pack2bf(float a, float b) {
    return (unsigned int)f2bf(a) | ((unsigned int)f2bf(b) << 16);
}

// order-preserving float <-> uint key (for atomicMax-based segment max)
static __device__ __forceinline__ unsigned int fkey(float f) {
    unsigned int b = __float_as_uint(f);
    return b ^ ((b >> 31) ? 0xFFFFFFFFu : 0x80000000u);
}
static __device__ __forceinline__ float keyf(unsigned int k) {
    unsigned int b = k ^ ((k >> 31) ? 0x80000000u : 0xFFFFFFFFu);
    return __uint_as_float(b);
}

static __device__ __forceinline__ float lrelu(float e) {
    return e > 0.f ? e : NEG_SLOPE * e;
}

// ---------------------------------------------------------------------------
// K1: h1 = x @ W1 via v_wmma_f32_16x16x32_bf16.
// Block = 128 threads (4 waves). Wave w owns output col tile n0 = w*16.
// B (W1) fragments are built once and kept in registers for the whole kernel;
// A tiles are staged per 16-node tile in fragment-swizzled order so each
// lane's 16 bf16 operand values are one contiguous 32B LDS run.
//
// Fragment layouts used here (16-bit operands, wave32):
//   A 16x32: lane l (r=l&15, hi=l>>4): element j -> K = hi*16 + j
//   B 32x16: lane l (n=l&15, hi=l>>4): element j -> K = hi*16 + j
//   D 16x16: VGPR v, lane l -> (M = v + hi*8, N = l&15)
// ---------------------------------------------------------------------------
__global__ void gat_gemm1_wmma(const float* __restrict__ x,
                               const float* __restrict__ W1,
                               float* __restrict__ h1,
                               int N, int nTiles) {
    // A tile, fragment-swizzled: [kstep(4)][lane(32)][elem(16)] bf16 = 4 KB
    __shared__ __align__(32) unsigned short lA[4 * 32 * 16];
    // B fragments, swizzled:    [wave(4)][kstep(4)][lane(32)][elem(16)] = 16 KB
    __shared__ __align__(32) unsigned short lB[4 * 4 * 32 * 16];

    const int tid  = threadIdx.x;   // 0..127
    const int wave = tid >> 5;      // 0..3 -> col tile n0 = wave*16
    const int lane = tid & 31;
    const int hi   = lane >> 4;

    // ---- stage W1 fragments (once per block), swizzled per-lane -----------
    for (int i = tid; i < 4 * 4 * 32 * 16; i += 128) {
        int w  = i >> 11;
        int s  = (i >> 9) & 3;
        int ln = (i >> 4) & 31;
        int j  = i & 15;
        int k  = s * 32 + (ln >> 4) * 16 + j;       // B: K = hi*16 + j
        int n  = w * 16 + (ln & 15);
        lB[i] = f2bf(W1[k * DOUT + n]);
    }
    __syncthreads();

    // ---- hoist this wave's B fragments into registers ---------------------
    const v16us* bPtr = (const v16us*)lB;
    v16bf bv[4];
#pragma unroll
    for (int s = 0; s < 4; ++s)
        bv[s] = __builtin_bit_cast(v16bf, bPtr[(wave * 4 + s) * 32 + lane]);

    const v16us* aPtr = (const v16us*)lA;

    for (int t = blockIdx.x; t < nTiles; t += gridDim.x) {
        const int node0 = t * 16;
        __syncthreads();   // protect lA from previous iteration's readers

        // ---- stage A tile: 256 half-fragments of 8 contiguous source floats.
        // Each thread: 2x (float4 x2 global load -> 8xbf16 pack -> ds b128).
        for (int u = tid; u < 256; u += 128) {
            int s    = u >> 6;
            int ln   = (u >> 1) & 31;
            int half = u & 1;
            int row  = node0 + (ln & 15);
            row = row < N ? row : N - 1;            // clamp ragged last tile
            int kb = s * 32 + (ln >> 4) * 16 + half * 8;
            const float4* px = (const float4*)&x[(size_t)row * DIN + kb];
            float4 f0 = px[0], f1 = px[1];
            uint4 pk;
            pk.x = pack2bf(f0.x, f0.y);
            pk.y = pack2bf(f0.z, f0.w);
            pk.z = pack2bf(f1.x, f1.y);
            pk.w = pack2bf(f1.z, f1.w);
            *(uint4*)&lA[(s * 32 + ln) * 16 + half * 8] = pk;
        }
        __syncthreads();

        // ---- load all A fragments, then issue WMMAs back-to-back ----------
        v16bf av[4];
#pragma unroll
        for (int s = 0; s < 4; ++s)
            av[s] = __builtin_bit_cast(v16bf, aPtr[s * 32 + lane]);

        v8f acc = {};
#pragma unroll
        for (int s = 0; s < 4; ++s)                 // K = 4 * 32 = 128
            acc = __builtin_amdgcn_wmma_f32_16x16x32_bf16(
                false, av[s], false, bv[s], (short)0, acc, false, false);

        // ---- store D: VGPR v, lane l -> (M = v + hi*8, Ncol = l&15) -------
        float* pD = h1 + (size_t)(node0 + hi * 8) * DOUT + wave * 16 + (lane & 15);
        if (node0 + 16 <= N) {                      // uniform: full tile
#pragma unroll
            for (int v = 0; v < 8; ++v) pD[v * DOUT] = acc[v];
        } else {                                    // ragged last tile
#pragma unroll
            for (int v = 0; v < 8; ++v)
                if (node0 + hi * 8 + v < N) pD[v * DOUT] = acc[v];
        }
    }
}

// ---------------------------------------------------------------------------
// K2: per-node attention dots for layer 1 + init of accumulators.
// ---------------------------------------------------------------------------
__global__ void gat_prep1(const float* __restrict__ h1,
                          const float* __restrict__ a_src,
                          const float* __restrict__ a_dst,
                          float* __restrict__ as1, float* __restrict__ ad1,
                          unsigned int* __restrict__ emaxKey,
                          float* __restrict__ denom,
                          float* __restrict__ agg, int N) {
    int n = blockIdx.x * blockDim.x + threadIdx.x;
    if (n >= N) return;
    const float* hn = h1 + (size_t)n * DOUT;
    float s[4] = {0.f, 0.f, 0.f, 0.f}, d[4] = {0.f, 0.f, 0.f, 0.f};
#pragma unroll
    for (int h = 0; h < 4; ++h)
#pragma unroll
        for (int c = 0; c < 16; ++c) {
            float v = hn[h * 16 + c];
            s[h] += v * a_src[h * 16 + c];
            d[h] += v * a_dst[h * 16 + c];
        }
#pragma unroll
    for (int h = 0; h < 4; ++h) {
        as1[n * 4 + h] = s[h];
        ad1[n * 4 + h] = d[h];
        emaxKey[n * 4 + h] = 0u;   // key-space -infinity
        denom[n * 4 + h] = 0.f;
    }
    float* an = agg + (size_t)n * DOUT;
#pragma unroll
    for (int c = 0; c < DOUT; ++c) an[c] = 0.f;
}

static __device__ __forceinline__ void edge_sd(const int* __restrict__ src,
                                               const int* __restrict__ dst,
                                               long long g, long long E,
                                               int& s, int& d) {
    if (g < E) { s = src[g]; d = dst[g]; }
    else       { s = (int)(g - E); d = s; }   // appended self-loops
}

// K3: segment max over incoming edges (4 heads) via ordered-key atomicMax.
__global__ void gat_edge_max1(const int* __restrict__ src, const int* __restrict__ dst,
                              const float* __restrict__ as1, const float* __restrict__ ad1,
                              unsigned int* __restrict__ emaxKey, long long E, int N) {
    long long g = (long long)blockIdx.x * blockDim.x + threadIdx.x;
    if (g >= E + N) return;
    int s, d; edge_sd(src, dst, g, E, s, d);
#pragma unroll
    for (int h = 0; h < 4; ++h) {
        float e = lrelu(as1[s * 4 + h] + ad1[d * 4 + h]);
        atomicMax(&emaxKey[d * 4 + h], fkey(e));
    }
}

// K4: denominators.
__global__ void gat_edge_sum1(const int* __restrict__ src, const int* __restrict__ dst,
                              const float* __restrict__ as1, const float* __restrict__ ad1,
                              const unsigned int* __restrict__ emaxKey,
                              float* __restrict__ denom, long long E, int N) {
    long long g = (long long)blockIdx.x * blockDim.x + threadIdx.x;
    if (g >= E + N) return;
    int s, d; edge_sd(src, dst, g, E, s, d);
#pragma unroll
    for (int h = 0; h < 4; ++h) {
        float e = lrelu(as1[s * 4 + h] + ad1[d * 4 + h]);
        atomicAdd(&denom[d * 4 + h], __expf(e - keyf(emaxKey[d * 4 + h])));
    }
}

// K5: weighted message aggregation. 64 threads per edge -> one channel each;
// adjacent lanes hit adjacent f32 of the same dst row (coalesced L2 atomics).
__global__ void gat_edge_agg1(const int* __restrict__ src, const int* __restrict__ dst,
                              const float* __restrict__ as1, const float* __restrict__ ad1,
                              const unsigned int* __restrict__ emaxKey,
                              const float* __restrict__ denom,
                              const float* __restrict__ h1,
                              float* __restrict__ agg, long long E, int N) {
    long long tg = (long long)blockIdx.x * blockDim.x + threadIdx.x;
    long long g  = tg >> 6;
    int c        = (int)(tg & 63);
    if (g >= E + N) return;
    int s, d; edge_sd(src, dst, g, E, s, d);
    int h = c >> 4;
    float e     = lrelu(as1[s * 4 + h] + ad1[d * 4 + h]);
    float alpha = __expf(e - keyf(emaxKey[d * 4 + h])) / denom[d * 4 + h];
    atomicAdd(&agg[(size_t)d * DOUT + c], h1[(size_t)s * DOUT + c] * alpha);
}

// K6: finalize layer1 (bias + relu), project 64->2, layer-2 att dots + init.
__global__ void gat_prep2(const float* __restrict__ agg1,
                          const float* __restrict__ b1,
                          const float* __restrict__ W2,
                          const float* __restrict__ a_src2,
                          const float* __restrict__ a_dst2,
                          float* __restrict__ h2,
                          float* __restrict__ as2, float* __restrict__ ad2,
                          unsigned int* __restrict__ emaxKey2,
                          float* __restrict__ denom2,
                          float* __restrict__ agg2, int N) {
    int n = blockIdx.x * blockDim.x + threadIdx.x;
    if (n >= N) return;
    const float* an = agg1 + (size_t)n * DOUT;
    float acc0 = 0.f, acc1 = 0.f;
#pragma unroll
    for (int c = 0; c < DOUT; ++c) {
        float v = an[c] + b1[c];
        v = v > 0.f ? v : 0.f;               // relu between layers
        acc0 += v * W2[c * 2 + 0];
        acc1 += v * W2[c * 2 + 1];
    }
    h2[n * 2 + 0] = acc0;
    h2[n * 2 + 1] = acc1;
    as2[n] = acc0 * a_src2[0] + acc1 * a_src2[1];
    ad2[n] = acc0 * a_dst2[0] + acc1 * a_dst2[1];
    emaxKey2[n] = 0u;
    denom2[n]   = 0.f;
    agg2[n * 2 + 0] = 0.f;
    agg2[n * 2 + 1] = 0.f;
}

__global__ void gat_edge_max2(const int* __restrict__ src, const int* __restrict__ dst,
                              const float* __restrict__ as2, const float* __restrict__ ad2,
                              unsigned int* __restrict__ emaxKey2, long long E, int N) {
    long long g = (long long)blockIdx.x * blockDim.x + threadIdx.x;
    if (g >= E + N) return;
    int s, d; edge_sd(src, dst, g, E, s, d);
    atomicMax(&emaxKey2[d], fkey(lrelu(as2[s] + ad2[d])));
}

__global__ void gat_edge_sum2(const int* __restrict__ src, const int* __restrict__ dst,
                              const float* __restrict__ as2, const float* __restrict__ ad2,
                              const unsigned int* __restrict__ emaxKey2,
                              float* __restrict__ denom2, long long E, int N) {
    long long g = (long long)blockIdx.x * blockDim.x + threadIdx.x;
    if (g >= E + N) return;
    int s, d; edge_sd(src, dst, g, E, s, d);
    float e = lrelu(as2[s] + ad2[d]);
    atomicAdd(&denom2[d], __expf(e - keyf(emaxKey2[d])));
}

__global__ void gat_edge_agg2(const int* __restrict__ src, const int* __restrict__ dst,
                              const float* __restrict__ as2, const float* __restrict__ ad2,
                              const unsigned int* __restrict__ emaxKey2,
                              const float* __restrict__ denom2,
                              const float* __restrict__ h2,
                              float* __restrict__ agg2, long long E, int N) {
    long long g = (long long)blockIdx.x * blockDim.x + threadIdx.x;
    if (g >= E + N) return;
    int s, d; edge_sd(src, dst, g, E, s, d);
    float e     = lrelu(as2[s] + ad2[d]);
    float alpha = __expf(e - keyf(emaxKey2[d])) / denom2[d];
    atomicAdd(&agg2[d * 2 + 0], h2[s * 2 + 0] * alpha);
    atomicAdd(&agg2[d * 2 + 1], h2[s * 2 + 1] * alpha);
}

// K10: out = agg2 + b2   (H=1 -> mean over heads is identity)
__global__ void gat_final(const float* __restrict__ agg2,
                          const float* __restrict__ b2,
                          float* __restrict__ out, int N) {
    int n = blockIdx.x * blockDim.x + threadIdx.x;
    if (n >= N) return;
    out[n * 2 + 0] = agg2[n * 2 + 0] + b2[0];
    out[n * 2 + 1] = agg2[n * 2 + 1] + b2[1];
}

// ---------------------------------------------------------------------------
extern "C" void kernel_launch(void* const* d_in, const int* in_sizes, int n_in,
                              void* d_out, int out_size, void* d_ws, size_t ws_size,
                              hipStream_t stream) {
    const float* x        = (const float*)d_in[0];
    const int*   ei       = (const int*)  d_in[1];   // [2, E] (src row, dst row)
    const float* W1       = (const float*)d_in[2];
    const float* att_src1 = (const float*)d_in[3];
    const float* att_dst1 = (const float*)d_in[4];
    const float* b1       = (const float*)d_in[5];
    const float* W2       = (const float*)d_in[6];
    const float* att_src2 = (const float*)d_in[7];
    const float* att_dst2 = (const float*)d_in[8];
    const float* b2       = (const float*)d_in[9];
    float* out = (float*)d_out;

    const int       N = in_sizes[0] / DIN;
    const long long E = in_sizes[1] / 2;
    const int* src = ei;
    const int* dst = ei + E;
    const long long Etot = E + N;

    // workspace carve-up (256B aligned)
    char* p = (char*)d_ws;
    size_t off = 0;
    auto carve = [&](size_t bytes) {
        void* q = p + off;
        off += (bytes + 255) & ~(size_t)255;
        return q;
    };
    float*        h1       = (float*)       carve((size_t)N * DOUT * 4);
    float*        as1      = (float*)       carve((size_t)N * 4 * 4);
    float*        ad1      = (float*)       carve((size_t)N * 4 * 4);
    unsigned int* emaxKey1 = (unsigned int*)carve((size_t)N * 4 * 4);
    float*        denom1   = (float*)       carve((size_t)N * 4 * 4);
    float*        agg1     = (float*)       carve((size_t)N * DOUT * 4);
    float*        h2       = (float*)       carve((size_t)N * 2 * 4);
    float*        as2      = (float*)       carve((size_t)N * 4);
    float*        ad2      = (float*)       carve((size_t)N * 4);
    unsigned int* emaxKey2 = (unsigned int*)carve((size_t)N * 4);
    float*        denom2   = (float*)       carve((size_t)N * 4);
    float*        agg2     = (float*)       carve((size_t)N * 2 * 4);
    (void)ws_size; (void)n_in; (void)out_size;

    const int nTiles = (N + 15) / 16;
    int gemmBlocks = nTiles < 1024 ? nTiles : 1024;
    gat_gemm1_wmma<<<gemmBlocks, 128, 0, stream>>>(x, W1, h1, N, nTiles);

    const int TB = 256;
    int nodeBlocks = (N + TB - 1) / TB;
    gat_prep1<<<nodeBlocks, TB, 0, stream>>>(h1, att_src1, att_dst1,
                                             as1, ad1, emaxKey1, denom1, agg1, N);

    long long edgeBlocks = (Etot + TB - 1) / TB;
    gat_edge_max1<<<(unsigned)edgeBlocks, TB, 0, stream>>>(src, dst, as1, ad1,
                                                           emaxKey1, E, N);
    gat_edge_sum1<<<(unsigned)edgeBlocks, TB, 0, stream>>>(src, dst, as1, ad1,
                                                           emaxKey1, denom1, E, N);
    long long aggBlocks = (Etot * 64 + TB - 1) / TB;
    gat_edge_agg1<<<(unsigned)aggBlocks, TB, 0, stream>>>(src, dst, as1, ad1,
                                                          emaxKey1, denom1, h1,
                                                          agg1, E, N);

    gat_prep2<<<nodeBlocks, TB, 0, stream>>>(agg1, b1, W2, att_src2, att_dst2,
                                             h2, as2, ad2, emaxKey2, denom2, agg2, N);
    gat_edge_max2<<<(unsigned)edgeBlocks, TB, 0, stream>>>(src, dst, as2, ad2,
                                                           emaxKey2, E, N);
    gat_edge_sum2<<<(unsigned)edgeBlocks, TB, 0, stream>>>(src, dst, as2, ad2,
                                                           emaxKey2, denom2, E, N);
    gat_edge_agg2<<<(unsigned)edgeBlocks, TB, 0, stream>>>(src, dst, as2, ad2,
                                                           emaxKey2, denom2, h2,
                                                           agg2, E, N);
    gat_final<<<nodeBlocks, TB, 0, stream>>>(agg2, b2, out, N);
}
